// GlobalContextVitStage_30124900614410
// MI455X (gfx1250) — matrix operand order repaired and checked by reading, not compiled, not run on hardware
//
#include <hip/hip_runtime.h>
#include <hip/hip_bf16.h>
#include <math.h>
#include <stdint.h>

// ---------------------------------------------------------------------------
// GCViT stage for MI455X (gfx1250, wave32).
// All matmul-heavy work runs through one bf16 WMMA GEMM kernel
// (v_wmma_f32_16x16x32_bf16, f32 accumulation). Full-tile GEMMs stage the A
// tile with CDNA5 async copies (global_load_async_to_lds_b128 + s_wait_asynccnt).
// ---------------------------------------------------------------------------

typedef __attribute__((ext_vector_type(16))) __bf16 v16bf;
typedef __attribute__((ext_vector_type(8)))  float  v8f;

union AFrag { v16bf v; uint4 q[2]; };
union BFrag { v16bf v; uint4 q[2]; };

#define BM 128
#define BN 128
#define BK 32
#define AST 48    // LDS row stride for A tile (bf16 elems, 96B = 16B multiple)
#define BST 144   // LDS row stride for B tile (bf16 elems, 288B = 16B multiple)

__device__ __forceinline__ float gelu_f(float x) {
    return 0.5f * x * (1.0f + erff(x * 0.70710678118654752f));
}

// ---------------------------------------------------------------------------
// Generic strided-batched GEMM:  C[m,n] = sum_k A[m,k] * B'[k,n]
//   btrans=1: B is [N,K] row-major (torch-style weight), B'[k,n] = B[n,k]
//   btrans=0: B is [K,N] row-major
// mode: 0 -> Cf = acc
//       1 -> Cb = bf16(acc + bias)
//       2 -> Cb = bf16(gelu(acc + bias))
//       3 -> Cf = acc + bias + resid
// fast=1 requires: M%128==0, N%128==0, K%32==0, btrans==1, lda%8==0, ldb%8==0,
// 16B-aligned A/B bases -> unguarded b128 loads, async A-tile staging.
// ---------------------------------------------------------------------------
__global__ __launch_bounds__(256)
void wmma_gemm(const __bf16* __restrict__ A, const __bf16* __restrict__ B,
               float* __restrict__ Cf, __bf16* __restrict__ Cb,
               const float* __restrict__ bias, const float* __restrict__ resid,
               int M, int N, int K, int lda, int ldb, int ldc,
               long long sAy, long long sAz, long long sBy, long long sBz,
               long long sCy, long long sCz,
               int btrans, int mode, int tilesN, int fast)
{
    __shared__ alignas(16) __bf16 As[BM * AST];
    __shared__ alignas(16) __bf16 Bs[BK * BST];

    const int tid = threadIdx.x;
    const int tm = (int)blockIdx.x / tilesN;
    const int tn = (int)blockIdx.x % tilesN;
    const int m0 = tm * BM, n0 = tn * BN;

    A += (long long)blockIdx.y * sAy + (long long)blockIdx.z * sAz;
    B += (long long)blockIdx.y * sBy + (long long)blockIdx.z * sBz;
    const long long coff = (long long)blockIdx.y * sCy + (long long)blockIdx.z * sCz;

    const int wv = tid >> 5, lane = tid & 31;
    const int wm = wv >> 2, wn = wv & 3;          // 2 x 4 wave grid
    const int half = lane >> 4, l16 = lane & 15;

    const unsigned asBase = (unsigned)(uintptr_t)(&As[0]);

    v8f acc[4][2] = {};
    const int kTiles = (K + BK - 1) / BK;

    for (int kt = 0; kt < kTiles; ++kt) {
        const int k0 = kt * BK;

        if (fast) {
            // ---- A tile via async DMA to LDS: 2 x b128 per thread ----
            #pragma unroll
            for (int it = 0; it < 2; ++it) {
                int g = tid + it * 256;
                int r = g >> 2, sg = g & 3;                 // row, 16B segment
                unsigned lds = asBase + (unsigned)(r * (AST * 2) + sg * 16);
                const __bf16* src = A + (long long)(m0 + r) * lda + k0 + sg * 8;
                asm volatile("global_load_async_to_lds_b128 %0, %1, off"
                             :: "v"(lds), "v"(src) : "memory");
            }
            // ---- B tile (weight [N,K]) -> k-major LDS, b128 global loads ----
            #pragma unroll
            for (int it = 0; it < 2; ++it) {
                int g = tid + it * 256;
                int r = g >> 2, ks = (g & 3) << 3;          // n row, k segment
                uint4 bv = *(const uint4*)(B + (long long)(n0 + r) * ldb + k0 + ks);
                const __bf16* t8 = (const __bf16*)&bv;
                #pragma unroll
                for (int j = 0; j < 8; ++j) Bs[(ks + j) * BST + r] = t8[j];
            }
            if (kt + 1 < kTiles)
                __builtin_prefetch(B + (long long)(n0 + (tid >> 1)) * ldb + (k0 + BK), 0, 1);
            asm volatile("s_wait_asynccnt 0x0" ::: "memory");
        } else {
            // ---- guarded path (branchless: clamped loads + selects) ----
            #pragma unroll
            for (int it = 0; it < 2; ++it) {
                int g = tid + it * 256;
                int r = g >> 2, ks = (g & 3) << 3;
                int gm = m0 + r, gk = k0 + ks;
                int cm = gm < M ? gm : M - 1;
                const __bf16* p = A + (long long)cm * lda;
                #pragma unroll
                for (int j = 0; j < 8; ++j) {
                    int kk = gk + j;
                    int ck = kk < K ? kk : K - 1;
                    __bf16 v = p[ck];
                    As[r * AST + ks + j] = (gm < M && kk < K) ? v : (__bf16)0.0f;
                }
            }
            if (btrans) {
                #pragma unroll
                for (int it = 0; it < 2; ++it) {
                    int g = tid + it * 256;
                    int r = g >> 2, ks = (g & 3) << 3;      // r = n, ks = k segment
                    int gn = n0 + r, gk = k0 + ks;
                    int cn = gn < N ? gn : N - 1;
                    const __bf16* p = B + (long long)cn * ldb;
                    #pragma unroll
                    for (int j = 0; j < 8; ++j) {
                        int kk = gk + j;
                        int ck = kk < K ? kk : K - 1;
                        __bf16 v = p[ck];
                        Bs[(ks + j) * BST + r] = (gn < N && kk < K) ? v : (__bf16)0.0f;
                    }
                }
            } else {
                #pragma unroll
                for (int it = 0; it < 2; ++it) {
                    int g = tid + it * 256;
                    int kk = g >> 4, ns = (g & 15) << 3;    // kk = k, ns = n segment
                    int gk = k0 + kk, gn = n0 + ns;
                    int ck = gk < K ? gk : K - 1;
                    const __bf16* p = B + (long long)ck * ldb;
                    #pragma unroll
                    for (int j = 0; j < 8; ++j) {
                        int nn = gn + j;
                        int cn = nn < N ? nn : N - 1;
                        __bf16 v = p[cn];
                        Bs[kk * BST + ns + j] = (gk < K && nn < N) ? v : (__bf16)0.0f;
                    }
                }
            }
        }

        __syncthreads();

        // ---- fragments + WMMA ----
        AFrag af[4];
        #pragma unroll
        for (int i = 0; i < 4; ++i) {
            const __bf16* pb = &As[(wm * 64 + i * 16 + l16) * AST];
            af[i].q[0] = *(const uint4*)(pb + 8 * half);        // K 0..7 (+8*half)
            af[i].q[1] = *(const uint4*)(pb + 16 + 8 * half);   // K 16..23 (+8*half)
        }
        BFrag bf[2];
        #pragma unroll
        for (int j = 0; j < 2; ++j) {
            const __bf16* pb = &Bs[wn * 32 + j * 16 + l16]; // N = l16, K 0..7/16..23 (+8*half)
            #pragma unroll
            for (int s = 0; s < 16; ++s)
                bf[j].v[s] = pb[(s < 8 ? 8 * half + s : 16 + 8 * half + (s - 8)) * BST];
        }
        #pragma unroll
        for (int i = 0; i < 4; ++i)
            #pragma unroll
            for (int j = 0; j < 2; ++j)
                acc[i][j] = __builtin_amdgcn_wmma_f32_16x16x32_bf16(
                    false, af[i].v, false, bf[j].v, (short)0, acc[i][j], false, false);

        __syncthreads();
    }

    // ---- epilogue ----
    #pragma unroll
    for (int i = 0; i < 4; ++i) {
        #pragma unroll
        for (int j = 0; j < 2; ++j) {
            int gmBase = m0 + wm * 64 + i * 16 + 8 * half;
            int gn = n0 + wn * 32 + j * 16 + l16;
            if (gn >= N) continue;
            float bv = bias ? bias[gn] : 0.0f;
            #pragma unroll
            for (int r = 0; r < 8; ++r) {
                int gm = gmBase + r;
                if (gm >= M) continue;
                long long off = coff + (long long)gm * ldc + gn;
                float v = acc[i][j][r];
                if (mode == 0)      Cf[off] = v;
                else if (mode == 1) Cb[off] = (__bf16)(v + bv);
                else if (mode == 2) Cb[off] = (__bf16)gelu_f(v + bv);
                else                Cf[off] = v + bv + resid[off];
            }
        }
    }
}

// ---------------------------------------------------------------------------
// Elementwise / normalization / conv helper kernels
// ---------------------------------------------------------------------------
__global__ void f2bf_kernel(const float* __restrict__ in, __bf16* __restrict__ out, long long n) {
    long long i = (long long)blockIdx.x * blockDim.x + threadIdx.x;
    if (i < n) out[i] = (__bf16)in[i];
}

// LayerNorm over channel dim of NCHW (ln2d); one thread per (b,h,w)
__global__ void ln2d_nchw_kernel(const float* __restrict__ in, const float* __restrict__ g,
                                 const float* __restrict__ b, float* __restrict__ out,
                                 int C, int HW, int total) {
    int p = blockIdx.x * blockDim.x + threadIdx.x;
    if (p >= total) return;
    int bb = p / HW, hw = p % HW;
    const float* base = in + (long long)bb * C * HW + hw;
    float s = 0.f, s2 = 0.f;
    for (int c = 0; c < C; ++c) { float v = base[(long long)c * HW]; s += v; s2 += v * v; }
    float m = s / C;
    float inv = rsqrtf(s2 / C - m * m + 1e-5f);
    float* ob = out + (long long)bb * C * HW + hw;
    for (int c = 0; c < C; ++c) {
        float v = base[(long long)c * HW];
        ob[(long long)c * HW] = (v - m) * inv * g[c] + b[c];
    }
}

// LayerNorm over last dim (rows x C); block per row
__global__ void ln_rows_kernel(const float* __restrict__ in, const float* __restrict__ g,
                               const float* __restrict__ b, float* __restrict__ outf,
                               __bf16* __restrict__ outb, int C) {
    __shared__ float r1[256];
    __shared__ float r2[256];
    long long row = blockIdx.x;
    const float* base = in + row * C;
    float s = 0.f, s2 = 0.f;
    for (int c = threadIdx.x; c < C; c += 256) { float v = base[c]; s += v; s2 += v * v; }
    r1[threadIdx.x] = s; r2[threadIdx.x] = s2; __syncthreads();
    for (int st = 128; st > 0; st >>= 1) {
        if (threadIdx.x < st) { r1[threadIdx.x] += r1[threadIdx.x + st]; r2[threadIdx.x] += r2[threadIdx.x + st]; }
        __syncthreads();
    }
    float m = r1[0] / C;
    float inv = rsqrtf(r2[0] / C - m * m + 1e-5f);
    for (int c = threadIdx.x; c < C; c += 256) {
        float v = (base[c] - m) * inv * g[c] + b[c];
        if (outf) outf[row * C + c] = v;
        if (outb) outb[row * C + c] = (__bf16)v;
    }
}

// depthwise 3x3, pad 1, stride 1, fused exact GELU (NCHW)
__global__ void dwconv3x3_gelu_kernel(const float* __restrict__ in, const float* __restrict__ w,
                                      float* __restrict__ out, int C, int H, int W, long long total) {
    long long i = (long long)blockIdx.x * blockDim.x + threadIdx.x;
    if (i >= total) return;
    int x = (int)(i % W); long long t = i / W;
    int y = (int)(t % H); t /= H;
    int c = (int)(t % C); int bb = (int)(t / C);
    const float* base = in + ((long long)bb * C + c) * H * W;
    const float* wc = w + c * 9;
    float acc = 0.f;
    #pragma unroll
    for (int kh = 0; kh < 3; ++kh) {
        int iy = y + kh - 1; if (iy < 0 || iy >= H) continue;
        #pragma unroll
        for (int kw = 0; kw < 3; ++kw) {
            int ix = x + kw - 1; if (ix < 0 || ix >= W) continue;
            acc += wc[kh * 3 + kw] * base[iy * W + ix];
        }
    }
    out[i] = gelu_f(acc);
}

// SE: global average pool, block per (b*C + c)
__global__ void se_pool_kernel(const float* __restrict__ in, float* __restrict__ s, int HW) {
    __shared__ float red[256];
    long long bc = blockIdx.x;
    const float* base = in + bc * HW;
    float acc = 0.f;
    for (int p = threadIdx.x; p < HW; p += 256) acc += base[p];
    red[threadIdx.x] = acc; __syncthreads();
    for (int st = 128; st > 0; st >>= 1) {
        if (threadIdx.x < st) red[threadIdx.x] += red[threadIdx.x + st];
        __syncthreads();
    }
    if (threadIdx.x == 0) s[bc] = red[0] / (float)HW;
}

// SE MLP: s1 = gelu(s @ w1^T), sv = sigmoid(s1 @ w2^T); block per batch
__global__ void se_mlp_kernel(const float* __restrict__ s, const float* __restrict__ w1,
                              const float* __restrict__ w2, float* __restrict__ sv, int C, int R) {
    __shared__ float ls[384];
    __shared__ float l1[96];
    int bb = blockIdx.x;
    for (int c = threadIdx.x; c < C; c += 256) ls[c] = s[bb * C + c];
    __syncthreads();
    for (int r = threadIdx.x; r < R; r += 256) {
        float a = 0.f;
        for (int c = 0; c < C; ++c) a += w1[r * C + c] * ls[c];
        l1[r] = gelu_f(a);
    }
    __syncthreads();
    for (int c = threadIdx.x; c < C; c += 256) {
        float a = 0.f;
        for (int r = 0; r < R; ++r) a += w2[c * R + r] * l1[r];
        sv[bb * C + c] = 1.0f / (1.0f + expf(-a));
    }
}

// out[b,o,:] = resid[b,o,:] + sum_c pw[o,c]*sv[b,c]*h[b,c,:]   (block per (b,o))
__global__ void pw_scale_add_kernel(const float* __restrict__ h, const float* __restrict__ sv,
                                    const float* __restrict__ pw, const float* __restrict__ resid,
                                    float* __restrict__ out, int C, int HW) {
    __shared__ float ws[384];
    int bo = blockIdx.x;
    int bb = bo / C, o = bo % C;
    for (int c = threadIdx.x; c < C; c += 256) ws[c] = pw[o * C + c] * sv[bb * C + c];
    __syncthreads();
    const float* hb = h + (long long)bb * C * HW;
    long long ob = (long long)bo * HW;
    for (int p = threadIdx.x; p < HW; p += 256) {
        float acc = 0.f;
        for (int c = 0; c < C; ++c) acc += ws[c] * hb[(long long)c * HW + p];
        out[ob + p] = resid[ob + p] + acc;
    }
}

__global__ void maxpool3x3s2_kernel(const float* __restrict__ in, float* __restrict__ out,
                                    int C, int H, int W, int OH, int OW, long long total) {
    long long i = (long long)blockIdx.x * blockDim.x + threadIdx.x;
    if (i >= total) return;
    int ox = (int)(i % OW); long long t = i / OW;
    int oy = (int)(t % OH); t /= OH;
    int c = (int)(t % C); int bb = (int)(t / C);
    const float* base = in + ((long long)bb * C + c) * H * W;
    float acc = -3.0e38f;
    #pragma unroll
    for (int kh = 0; kh < 3; ++kh) {
        int iy = 2 * oy - 1 + kh; if (iy < 0 || iy >= H) continue;
        #pragma unroll
        for (int kw = 0; kw < 3; ++kw) {
            int ix = 2 * ox - 1 + kw; if (ix < 0 || ix >= W) continue;
            acc = fmaxf(acc, base[iy * W + ix]);
        }
    }
    out[i] = acc;
}

// im2col for 3x3 stride2 pad1 conv, NCHW input -> [B*OH*OW, C*9] bf16
__global__ void im2col_s2_kernel(const float* __restrict__ in, __bf16* __restrict__ out,
                                 int C, int H, int W, int OH, int OW, long long total) {
    long long i = (long long)blockIdx.x * blockDim.x + threadIdx.x;
    if (i >= total) return;
    int K9 = C * 9;
    int k = (int)(i % K9); long long r = i / K9;
    int c = k / 9, t = k % 9, kh = t / 3, kw = t % 3;
    int ox = (int)(r % OW); long long r2 = r / OW;
    int oy = (int)(r2 % OH); int bb = (int)(r2 / OH);
    int iy = 2 * oy - 1 + kh, ix = 2 * ox - 1 + kw;
    float v = (iy >= 0 && iy < H && ix >= 0 && ix < W)
                  ? in[(((long long)bb * C + c) * H + iy) * W + ix] : 0.f;
    out[i] = (__bf16)v;
}

// NHWC -> NCHW (thread per NCHW element)
__global__ void nhwc_to_nchw_kernel(const float* __restrict__ in, float* __restrict__ out,
                                    int C, int H, int W, long long total) {
    long long i = (long long)blockIdx.x * blockDim.x + threadIdx.x;
    if (i >= total) return;
    int x = (int)(i % W); long long t = i / W;
    int y = (int)(t % H); t /= H;
    int c = (int)(t % C); int bb = (int)(t / C);
    out[i] = in[(((long long)bb * H + y) * W + x) * C + c];
}

// NHWC [B,28,28,384] -> window-token order [(b*4+wh*2+ww)*196 + i*14+j, c]
__global__ void nhwc_to_win_kernel(const float* __restrict__ in, float* __restrict__ out,
                                   long long total) {
    long long idx = (long long)blockIdx.x * blockDim.x + threadIdx.x;
    if (idx >= total) return;
    int c = (int)(idx % 384); long long r = idx / 384;
    int n = (int)(r % 196); long long w = r / 196;
    int ww = (int)(w % 2); long long t = w / 2;
    int wh = (int)(t % 2); int bb = (int)(t / 2);
    int ii = n / 14, jj = n % 14;
    int y = wh * 14 + ii, x = ww * 14 + jj;
    out[idx] = in[(((long long)bb * 28 + y) * 28 + x) * 384 + c];
}

// window-token order -> NCHW output
__global__ void win_to_nchw_kernel(const float* __restrict__ in, float* __restrict__ out,
                                   long long total) {
    long long i = (long long)blockIdx.x * blockDim.x + threadIdx.x;
    if (i >= total) return;
    int x = (int)(i % 28); long long t = i / 28;
    int y = (int)(t % 28); t /= 28;
    int c = (int)(t % 384); int bb = (int)(t / 384);
    int wh = y / 14, ii = y % 14, ww = x / 14, jj = x % 14;
    long long row = ((long long)(bb * 4 + wh * 2 + ww)) * 196 + ii * 14 + jj;
    out[i] = in[row * 384 + c];
}

// NCHW [B,C,HW] -> token-major bf16 [B*HW, C]
__global__ void nchw_to_tok_bf16_kernel(const float* __restrict__ in, __bf16* __restrict__ out,
                                        int C, int HW, long long total) {
    long long i = (long long)blockIdx.x * blockDim.x + threadIdx.x;
    if (i >= total) return;
    int c = (int)(i % C); long long r = i / C;
    int p = (int)(r % HW); int bb = (int)(r / HW);
    out[i] = (__bf16)in[((long long)bb * C + c) * HW + p];
}

// softmax(S*scale + relbias[h]) over rows of 196; grid.x = nBatch*196
__global__ void attn_softmax_kernel(const float* __restrict__ S, __bf16* __restrict__ P,
                                    const float* __restrict__ table, float scale) {
    __shared__ float red[256];
    int bz = blockIdx.x / 196;
    int m = blockIdx.x % 196;
    int h = bz & 7;
    long long off = (long long)bz * 38416 + (long long)m * 196;
    int t = threadIdx.x;
    int im = m / 14, jm = m % 14;
    float val = -3.0e38f;
    if (t < 196) {
        int in_ = t / 14, jn = t % 14;
        int idx = (im - in_ + 13) * 27 + (jm - jn + 13);
        val = S[off + t] * scale + table[idx * 8 + h];
    }
    red[t] = val; __syncthreads();
    for (int st = 128; st > 0; st >>= 1) {
        if (t < st) red[t] = fmaxf(red[t], red[t + st]);
        __syncthreads();
    }
    float mx = red[0]; __syncthreads();
    float e = (t < 196) ? expf(val - mx) : 0.f;
    red[t] = e; __syncthreads();
    for (int st = 128; st > 0; st >>= 1) {
        if (t < st) red[t] += red[t + st];
        __syncthreads();
    }
    float inv = 1.0f / red[0];
    if (t < 196) P[off + t] = (__bf16)(e * inv);
}

// ---------------------------------------------------------------------------
// Host side
// ---------------------------------------------------------------------------
#define CDIV(a, b) (((a) + (b) - 1) / (b))

static inline void gemm(hipStream_t st, const __bf16* A, const __bf16* B,
                        float* Cf, __bf16* Cb, const float* bias, const float* resid,
                        int M, int N, int K, int lda, int ldb, int ldc,
                        long long sAy, long long sAz, long long sBy, long long sBz,
                        long long sCy, long long sCz, int btrans, int mode,
                        int gy, int gz) {
    int tM = CDIV(M, BM), tN = CDIV(N, BN);
    int fast = ((M & 127) == 0) && ((N & 127) == 0) && ((K & 31) == 0) &&
               (btrans == 1) && ((lda & 7) == 0) && ((ldb & 7) == 0);
    dim3 grid(tM * tN, gy, gz);
    wmma_gemm<<<grid, 256, 0, st>>>(A, B, Cf, Cb, bias, resid, M, N, K, lda, ldb, ldc,
                                    sAy, sAz, sBy, sBz, sCy, sCz, btrans, mode, tN, fast);
}

extern "C" void kernel_launch(void* const* d_in, const int* in_sizes, int n_in,
                              void* d_out, int out_size, void* d_ws, size_t ws_size,
                              hipStream_t stream) {
    (void)in_sizes; (void)n_in; (void)out_size; (void)ws_size;

    // ---- inputs (setup_inputs dict order) ----
    const float* x_in   = (const float*)d_in[0];   // [32,192,56,56]
    const float* ln1g   = (const float*)d_in[1];
    const float* ln1b   = (const float*)d_in[2];
    const float* ds_dw  = (const float*)d_in[3];   // [192,1,3,3]
    const float* ds_se1 = (const float*)d_in[4];   // [48,192]
    const float* ds_se2 = (const float*)d_in[5];   // [192,48]
    const float* ds_pw  = (const float*)d_in[6];   // [192,192]
    const float* red_w  = (const float*)d_in[7];   // [384,192,3,3]
    const float* ln2g   = (const float*)d_in[8];
    const float* ln2b   = (const float*)d_in[9];
    const float* gb_dw  = (const float*)d_in[10];  // [384,1,3,3]
    const float* gb_se1 = (const float*)d_in[11];  // [96,384]
    const float* gb_se2 = (const float*)d_in[12];  // [384,96]
    const float* gb_pw  = (const float*)d_in[13];  // [384,384]
    // blocks: 13 params each starting at 14 and 27
    const int b0 = 14, b1 = 27;

    float* out = (float*)d_out;                    // [32,384,28,28]

    // ---- workspace arena ----
    char* wsb = (char*)d_ws;
    size_t cur = 0;
    auto arena = [&](size_t bytes) -> void* {
        void* p = wsb + cur;
        cur = (cur + bytes + 255) & ~(size_t)255;
        return p;
    };
    const long long NT = 25088;          // tokens at 28x28 (= 128 windows * 196)
    void* R1  = arena((size_t)NT * 1536 * 2);          // xn/h2 f32 (77MB)  | fc1 bf16
    void* R2  = arena((size_t)32 * 192 * 3136 * 4);    // h1 / gb-dw / S-chunk f32
    void* R3  = arena((size_t)NT * 1728 * 2);          // im2col bf16 | qkv bf16
    void* R4  = arena((size_t)NT * 384 * 4);           // x28 nhwc f32 | P-chunk bf16
    void* R5  = arena((size_t)NT * 384 * 4);           // x28 nchw / g / x2 f32
    void* R6  = arena((size_t)NT * 384 * 4);           // xw (residual stream) f32
    void* R7  = arena((size_t)NT * 384 * 2);           // y/z bf16
    void* R8  = arena((size_t)NT * 384 * 2);           // attention output bf16
    void* R9a = arena((size_t)32 * 384 * 196 * 4);     // gq nchw f32
    void* R9b = arena((size_t)32 * 196 * 384 * 2);     // gq tokens bf16
    void* Rse = arena((size_t)(32 * 384 * 2) * 4);     // SE pooled + sigmoid scale
    void* Rw  = arena((size_t)4055040 * 2);            // all bf16 weights

    float*  xn     = (float*)R1;
    __bf16* fc1buf = (__bf16*)R1;
    float*  h1     = (float*)R2;
    float*  Sbuf   = (float*)R2;
    __bf16* cols   = (__bf16*)R3;
    __bf16* qkvbf  = (__bf16*)R3;
    float*  x28    = (float*)R4;
    __bf16* Pbuf   = (__bf16*)R4;
    float*  xnchw  = (float*)R5;
    float*  x2     = (float*)R5;
    float*  xw     = (float*)R6;
    __bf16* ybf    = (__bf16*)R7;
    __bf16* obf    = (__bf16*)R8;
    float*  gqnchw = (float*)R9a;
    __bf16* gqbf   = (__bf16*)R9b;
    float*  sbuf   = (float*)Rse;
    float*  svbuf  = sbuf + 32 * 384;

    // bf16 weight slices
    __bf16* wb = (__bf16*)Rw;
    size_t wcur = 0;
    auto wslice = [&](size_t n) -> __bf16* { __bf16* p = wb + wcur; wcur += n; return p; };
    __bf16* redw_bf  = wslice((size_t)384 * 1728);
    __bf16* qkvw0_bf = wslice((size_t)1152 * 384);
    __bf16* projw0_bf = wslice((size_t)384 * 384);
    __bf16* fc1w0_bf = wslice((size_t)1536 * 384);
    __bf16* fc2w0_bf = wslice((size_t)384 * 1536);
    __bf16* qkvw1_bf = wslice((size_t)768 * 384);
    __bf16* projw1_bf = wslice((size_t)384 * 384);
    __bf16* fc1w1_bf = wslice((size_t)1536 * 384);
    __bf16* fc2w1_bf = wslice((size_t)384 * 1536);

    auto cvt = [&](const float* src, __bf16* dst, long long n) {
        f2bf_kernel<<<(unsigned)CDIV(n, 256), 256, 0, stream>>>(src, dst, n);
    };
    cvt(red_w, redw_bf, 384LL * 1728);
    cvt((const float*)d_in[b0 + 2], qkvw0_bf, 1152LL * 384);
    cvt((const float*)d_in[b0 + 5], projw0_bf, 384LL * 384);
    cvt((const float*)d_in[b0 + 9], fc1w0_bf, 1536LL * 384);
    cvt((const float*)d_in[b0 + 11], fc2w0_bf, 384LL * 1536);
    cvt((const float*)d_in[b1 + 2], qkvw1_bf, 768LL * 384);
    cvt((const float*)d_in[b1 + 5], projw1_bf, 384LL * 384);
    cvt((const float*)d_in[b1 + 9], fc1w1_bf, 1536LL * 384);
    cvt((const float*)d_in[b1 + 11], fc2w1_bf, 384LL * 1536);

    // ================= downsample =================
    // ln2d over 192ch NCHW
    ln2d_nchw_kernel<<<CDIV(32 * 3136, 256), 256, 0, stream>>>(x_in, ln1g, ln1b, xn, 192, 3136, 32 * 3136);
    // MBConv @192
    dwconv3x3_gelu_kernel<<<CDIV(19267584LL, 256), 256, 0, stream>>>(xn, ds_dw, h1, 192, 56, 56, 19267584LL);
    se_pool_kernel<<<32 * 192, 256, 0, stream>>>(h1, sbuf, 3136);
    se_mlp_kernel<<<32, 256, 0, stream>>>(sbuf, ds_se1, ds_se2, svbuf, 192, 48);
    pw_scale_add_kernel<<<32 * 192, 256, 0, stream>>>(h1, svbuf, ds_pw, xn, xn, 192, 3136);
    // 3x3/s2 conv as im2col + WMMA GEMM -> NHWC f32
    im2col_s2_kernel<<<CDIV(43352064LL, 256), 256, 0, stream>>>(xn, cols, 192, 56, 56, 28, 28, 43352064LL);
    gemm(stream, cols, redw_bf, x28, nullptr, nullptr, nullptr,
         25088, 384, 1728, 1728, 1728, 384, 0, 0, 0, 0, 0, 0, 1, 0, 1, 1);
    // ln2d == LN over last dim in NHWC (in place)
    ln_rows_kernel<<<25088, 256, 0, stream>>>(x28, ln2g, ln2b, x28, nullptr, 384);

    // split into NCHW (for MBConv path) and window-token order (transformer stream)
    nhwc_to_nchw_kernel<<<CDIV(9633792LL, 256), 256, 0, stream>>>(x28, xnchw, 384, 28, 28, 9633792LL);
    nhwc_to_win_kernel<<<CDIV(9633792LL, 256), 256, 0, stream>>>(x28, xw, 9633792LL);

    // ================= global query =================
    dwconv3x3_gelu_kernel<<<CDIV(9633792LL, 256), 256, 0, stream>>>(xnchw, gb_dw, h1, 384, 28, 28, 9633792LL);
    se_pool_kernel<<<32 * 384, 256, 0, stream>>>(h1, sbuf, 784);
    se_mlp_kernel<<<32, 256, 0, stream>>>(sbuf, gb_se1, gb_se2, svbuf, 384, 96);
    pw_scale_add_kernel<<<32 * 384, 256, 0, stream>>>(h1, svbuf, gb_pw, xnchw, xnchw, 384, 784);
    maxpool3x3s2_kernel<<<CDIV(2408448LL, 256), 256, 0, stream>>>(xnchw, gqnchw, 384, 28, 28, 14, 14, 2408448LL);
    nchw_to_tok_bf16_kernel<<<CDIV(2408448LL, 256), 256, 0, stream>>>(gqnchw, gqbf, 384, 196, 2408448LL);

    // ================= transformer blocks =================
    const float scale = 0.1443375672974065f;  // 48^-0.5
    for (int blk = 0; blk < 2; ++blk) {
        const int pb = blk ? b1 : b0;
        const int useg = blk;                     // block1 uses global q
        const int ldq = useg ? 768 : 1152;
        const int koff = useg ? 0 : 384;
        const int voff = useg ? 384 : 768;
        const __bf16* qkvw = useg ? qkvw1_bf : qkvw0_bf;
        const __bf16* projw = useg ? projw1_bf : projw0_bf;
        const __bf16* fc1w = useg ? fc1w1_bf : fc1w0_bf;
        const __bf16* fc2w = useg ? fc2w1_bf : fc2w0_bf;
        const float* n1g = (const float*)d_in[pb + 0];
        const float* n1b = (const float*)d_in[pb + 1];
        const float* qkvb = (const float*)d_in[pb + 3];
        const float* table = (const float*)d_in[pb + 4];
        const float* projb = (const float*)d_in[pb + 6];
        const float* n2g = (const float*)d_in[pb + 7];
        const float* n2b = (const float*)d_in[pb + 8];
        const float* fc1b = (const float*)d_in[pb + 10];
        const float* fc2b = (const float*)d_in[pb + 12];

        // LN1 -> bf16 tokens
        ln_rows_kernel<<<25088, 256, 0, stream>>>(xw, n1g, n1b, nullptr, ybf, 384);
        // QKV (or KV) projection
        gemm(stream, ybf, qkvw, nullptr, qkvbf, qkvb, nullptr,
             25088, ldq, 384, 384, 384, ldq, 0, 0, 0, 0, 0, 0, 1, 1, 1, 1);

        // attention in 4 chunks of 32 windows (256 (window,head) batches each)
        for (int ch = 0; ch < 4; ++ch) {
            const long long chq = (long long)ch * 32 * 196 * ldq;
            const __bf16* qptr = useg ? gqbf : (qkvbf + chq);       // block1: q index = w % 32
            const int qld = useg ? 384 : 1152;
            // S = Q K^T   [196x196, K=48] per (window, head)
            gemm(stream, qptr, qkvbf + chq + koff, Sbuf, nullptr, nullptr, nullptr,
                 196, 196, 48, qld, ldq, 196,
                 (long long)196 * qld, 48, (long long)196 * ldq, 48,
                 8LL * 38416, 38416, 1, 0, 32, 8);
            // softmax(S*scale + relbias) -> bf16 P
            attn_softmax_kernel<<<256 * 196, 256, 0, stream>>>(Sbuf, Pbuf, table, scale);
            // O = P V   [196x48, K=196]
            gemm(stream, Pbuf, qkvbf + chq + voff, nullptr,
                 obf + (long long)ch * 32 * 196 * 384, nullptr, nullptr,
                 196, 48, 196, 196, ldq, 384,
                 8LL * 38416, 38416, (long long)196 * ldq, 48,
                 (long long)196 * 384, 48, 0, 1, 32, 8);
        }

        // x2 = xw + proj(o)
        gemm(stream, obf, projw, x2, nullptr, projb, xw,
             25088, 384, 384, 384, 384, 384, 0, 0, 0, 0, 0, 0, 1, 3, 1, 1);
        // LN2 -> bf16
        ln_rows_kernel<<<25088, 256, 0, stream>>>(x2, n2g, n2b, nullptr, ybf, 384);
        // fc1 + GELU -> bf16
        gemm(stream, ybf, fc1w, nullptr, fc1buf, fc1b, nullptr,
             25088, 1536, 384, 384, 384, 1536, 0, 0, 0, 0, 0, 0, 1, 2, 1, 1);
        // xw = x2 + fc2(.)
        gemm(stream, fc1buf, fc2w, xw, nullptr, fc2b, x2,
             25088, 384, 1536, 1536, 1536, 384, 0, 0, 0, 0, 0, 0, 1, 3, 1, 1);
    }

    // window-token order -> NCHW output
    win_to_nchw_kernel<<<CDIV(9633792LL, 256), 256, 0, stream>>>(xw, out, 9633792LL);
}
